// DeepGATNet_1133871366390
// MI455X (gfx1250) — matrix operand order repaired and checked by reading, not compiled, not run on hardware
//
#include <hip/hip_runtime.h>
#include <math.h>

// ---------------- problem constants (from reference) ----------------
#define NNODES 30000
#define NEDGES 480000
#define EE     (NEDGES + NNODES)   // edges + appended self loops
#define FIN    128
#define CH     64
#define HEADS  4
#define HC     (HEADS * CH)        // 256
#define NEG_SLOPE 0.2f
#define LN_EPS 1e-5f

typedef __attribute__((ext_vector_type(16))) _Float16 v16h;
typedef __attribute__((ext_vector_type(8)))  _Float16 v8h;
typedef __attribute__((ext_vector_type(8)))  float    v8f;
typedef __attribute__((ext_vector_type(4)))  float    v4f;

__device__ __forceinline__ float leaky(float x) { return x > 0.f ? x : NEG_SLOPE * x; }
__device__ __forceinline__ float eluf(float x)  { return x > 0.f ? x : (__expf(x) - 1.f); }

// order-preserving float<->uint transform for atomicMax on floats
__device__ __forceinline__ unsigned f2ord(float f) {
    unsigned u = __float_as_uint(f);
    return (u & 0x80000000u) ? ~u : (u | 0x80000000u);
}
__device__ __forceinline__ float ord2f(unsigned u) {
    return __uint_as_float((u & 0x80000000u) ? (u & 0x7fffffffu) : ~u);
}
#define ORD_NEG_INF 0x007FFFFFu   // f2ord(-inf)

// ---------------------------------------------------------------------------
// Templated WMMA GEMM:  C[M,N] = act( A[M,K] @ B[K,N] + bias[N] )
//   - compile-time N, K, ACT(0=none,1=ELU), HAS_BIAS -> fully unrolled K loop,
//     no runtime branches, shift-based addressing.
//   - B panel (NP<=64 cols, all K rows) staged once per block in LDS as f16
//     (col-major, KP=K+8 halves padding -> conflict-free ds_load_b128).
//   - A fragments: 4x global_load_b128 (v4f) + v_cvt_pk_f16_f32.
//   - 128 threads = 4 waves; wave w owns the 16x16 tile (blockIdx.x*16,
//     blockIdx.y*64 + w*16). Requires M%16==0 (M=30000), K%32==0.
// ---------------------------------------------------------------------------
template<int N, int K, int ACT, bool HAS_BIAS>
__global__ __launch_bounds__(128)
void wmma_gemm(const float* __restrict__ A, const float* __restrict__ B,
               const float* __restrict__ bias, float* __restrict__ C)
{
    constexpr int NP = (N < 64) ? N : 64;    // panel width handled per block
    constexpr int KP = K + 8;                // LDS row pitch (halves)
    __shared__ _Float16 ldsB[NP * KP];

    const int m0 = blockIdx.x * 16;
    const int n0 = blockIdx.y * 64;

    // ---- stage B panel to LDS as f16 (coalesced global reads across cols) --
    for (int idx = threadIdx.x; idx < NP * K; idx += 128) {
        int c = idx % NP, k = idx / NP;
        ldsB[c * KP + k] = (_Float16)B[(size_t)k * N + (n0 + c)];
    }
    __syncthreads();

    const int lane = threadIdx.x & 31;
    const int wave = threadIdx.x >> 5;
    const int nw   = wave * 16;              // tile col offset within panel
    if (NP < 64 && nw >= NP) return;         // idle waves for N=32
    const int hv   = lane >> 4;              // half-wave select
    const int l15  = lane & 15;

    const float* arow = A + (size_t)(m0 + l15) * K;
    const _Float16* bcol = &ldsB[(nw + l15) * KP];

    v8f c = {};
    #pragma unroll
    for (int k0 = 0; k0 < K; k0 += 32) {
        // A fragment 16x32: elems 0-7 -> K=k0+hv*8+j, elems 8-15 -> +16
        const v4f* ap4 = (const v4f*)(arow + k0 + hv * 8);
        v4f a0 = ap4[0], a1 = ap4[1], a2 = ap4[4], a3 = ap4[5];
        v16h a;
        #pragma unroll
        for (int j = 0; j < 4; ++j) {
            a[j]      = (_Float16)a0[j];
            a[j + 4]  = (_Float16)a1[j];
            a[j + 8]  = (_Float16)a2[j];
            a[j + 12] = (_Float16)a3[j];
        }
        // B fragment 32x16 from LDS: col = nw+l15, elems 0-7 -> K=k0+hv*8+j, elems 8-15 -> +16
        const v8h* bp = (const v8h*)(bcol + k0 + hv * 8);
        v8h blo = bp[0], bhi = bp[2];
        v16h b = __builtin_shufflevector(blo, bhi, 0, 1, 2, 3, 4, 5, 6, 7,
                                         8, 9, 10, 11, 12, 13, 14, 15);
        c = __builtin_amdgcn_wmma_f32_16x16x32_f16(
                false, a, false, b, (short)0, c, false, false);
    }

    const int  cn = n0 + nw + l15;
    const float bv = HAS_BIAS ? bias[cn] : 0.f;
    float* crow = C + (size_t)(m0 + hv * 8) * N + cn;
    #pragma unroll
    for (int r = 0; r < 8; ++r) {            // row = m0 + r + hv*8
        float v = c[r] + bv;
        if (ACT == 1) v = eluf(v);
        crow[(size_t)r * N] = v;
    }
}

// ---------------- simple fills (acc / segment state init) ----------------
__global__ void fill_f32(float* __restrict__ p, float v, int n) {
    int i = blockIdx.x * blockDim.x + threadIdx.x;
    if (i < n) p[i] = v;
}
__global__ void fill_u32(unsigned* __restrict__ p, unsigned v, int n) {
    int i = blockIdx.x * blockDim.x + threadIdx.x;
    if (i < n) p[i] = v;
}

// ---------------- per-(node,head) attention logits ----------------
template<int H>
__global__ void attn_node(const float* __restrict__ xl, const float* __restrict__ asrc,
                          const float* __restrict__ adst, float* __restrict__ es,
                          float* __restrict__ ed)
{
    int idx = blockIdx.x * blockDim.x + threadIdx.x;
    if (idx >= NNODES * H) return;
    int n = idx / H, hh = idx % H;
    const float* xr = xl + (size_t)n * H * CH + hh * CH;
    const float* av = asrc + hh * CH;
    const float* bv = adst + hh * CH;
    float s = 0.f, d = 0.f;
    #pragma unroll
    for (int c = 0; c < CH; ++c) { float v = xr[c]; s += v * av[c]; d += v * bv[c]; }
    es[idx] = s; ed[idx] = d;
}

__device__ __forceinline__ void edge_sd(const long long* __restrict__ ei, int i,
                                        int& s, int& d)
{
    if (i < NEDGES) { s = (int)ei[i]; d = (int)ei[NEDGES + i]; }
    else            { s = d = i - NEDGES; }          // appended self loops
}

// pass 1: per-dst segment max (numerically stable softmax)
template<int H>
__global__ void edge_max(const long long* __restrict__ ei, const float* __restrict__ es,
                         const float* __restrict__ ed, unsigned* __restrict__ mseg)
{
    int i = blockIdx.x * blockDim.x + threadIdx.x;
    if (i >= EE) return;
    int s, d; edge_sd(ei, i, s, d);
    #pragma unroll
    for (int hh = 0; hh < H; ++hh) {
        float e = leaky(es[s * H + hh] + ed[d * H + hh]);
        atomicMax(&mseg[d * H + hh], f2ord(e));
    }
}

// pass 2: exp(e - max) and per-dst sum
template<int H>
__global__ void edge_sum(const long long* __restrict__ ei, const float* __restrict__ es,
                         const float* __restrict__ ed, const unsigned* __restrict__ mseg,
                         float* __restrict__ sseg, float* __restrict__ ealpha)
{
    int i = blockIdx.x * blockDim.x + threadIdx.x;
    if (i >= EE) return;
    int s, d; edge_sd(ei, i, s, d);
    #pragma unroll
    for (int hh = 0; hh < H; ++hh) {
        float e  = leaky(es[s * H + hh] + ed[d * H + hh]);
        float ex = __expf(e - ord2f(mseg[d * H + hh]));
        ealpha[(size_t)i * H + hh] = ex;
        atomicAdd(&sseg[d * H + hh], ex);
    }
}

// pass 3: acc[dst] += alpha * xl[src]; one edge per block, thread t -> (h,c)
template<int H>
__global__ void edge_msg(const long long* __restrict__ ei, const float* __restrict__ xl,
                         const float* __restrict__ ealpha, const float* __restrict__ sseg,
                         float* __restrict__ acc)
{
    int i = blockIdx.x;
    int t = threadIdx.x;                  // t < H*64
    int s, d; edge_sd(ei, i, s, d);
    int hh = t >> 6;
    float alpha = ealpha[(size_t)i * H + hh] / sseg[d * H + hh];
    atomicAdd(&acc[(size_t)d * H * CH + t], alpha * xl[(size_t)s * H * CH + t]);
}

// ---------------- head-mean + bias + LayerNorm + residual (wave per node) ----
__global__ __launch_bounds__(256)
void mean_ln_res(const float* __restrict__ acc, const float* __restrict__ bias,
                 const float* __restrict__ g, const float* __restrict__ bln,
                 const float* __restrict__ res, float* __restrict__ hout)
{
    int wave = threadIdx.x >> 5, lane = threadIdx.x & 31;
    int node = blockIdx.x * 8 + wave;
    if (node >= NNODES) return;
    const float* a = acc + (size_t)node * HC;
    int c0 = lane, c1 = lane + 32;
    float v0 = 0.f, v1 = 0.f;
    #pragma unroll
    for (int hh = 0; hh < HEADS; ++hh) { v0 += a[hh * CH + c0]; v1 += a[hh * CH + c1]; }
    v0 = v0 * 0.25f + bias[c0];
    v1 = v1 * 0.25f + bias[c1];
    float sum = v0 + v1;
    #pragma unroll
    for (int off = 16; off > 0; off >>= 1) sum += __shfl_xor(sum, off, 32);
    float mu = sum * (1.f / 64.f);
    float d0 = v0 - mu, d1 = v1 - mu;
    float vs = d0 * d0 + d1 * d1;
    #pragma unroll
    for (int off = 16; off > 0; off >>= 1) vs += __shfl_xor(vs, off, 32);
    float inv = rsqrtf(vs * (1.f / 64.f) + LN_EPS);
    const float* r = res + (size_t)node * CH;
    float r0 = r[c0], r1 = r[c1];          // load residual before (possibly in-place) store
    float* o = hout + (size_t)node * CH;
    o[c0] = d0 * inv * g[c0] + bln[c0] + r0;
    o[c1] = d1 * inv * g[c1] + bln[c1] + r1;
}

// layer-3 finalize: h = acc + b3 (heads=1, no LN, no residual)
__global__ void add_bias64(const float* __restrict__ acc, const float* __restrict__ b,
                           float* __restrict__ h)
{
    int i = blockIdx.x * blockDim.x + threadIdx.x;
    if (i < NNODES * CH) h[i] = acc[i] + b[i & 63];
}

// MLP tail: out[n] = sigmoid(t2[n,:32] . w + b)
__global__ void mlp_final(const float* __restrict__ t2, const float* __restrict__ w,
                          const float* __restrict__ b, float* __restrict__ out)
{
    int n = blockIdx.x * blockDim.x + threadIdx.x;
    if (n >= NNODES) return;
    float s = b[0];
    #pragma unroll
    for (int j = 0; j < 32; ++j) s += t2[(size_t)n * 32 + j] * w[j];
    out[n] = 1.f / (1.f + __expf(-s));
}

// ---------------------------------------------------------------------------
extern "C" void kernel_launch(void* const* d_in, const int* in_sizes, int n_in,
                              void* d_out, int out_size, void* d_ws, size_t ws_size,
                              hipStream_t stream)
{
    (void)in_sizes; (void)n_in; (void)out_size; (void)ws_size;

    const float*     x    = (const float*)d_in[0];
    const long long* ei   = (const long long*)d_in[1];
    const float* W1   = (const float*)d_in[2];
    const float* as1  = (const float*)d_in[3];
    const float* ad1  = (const float*)d_in[4];
    const float* b1   = (const float*)d_in[5];
    const float* ln1g = (const float*)d_in[6];
    const float* ln1b = (const float*)d_in[7];
    const float* resW = (const float*)d_in[8];
    const float* resb = (const float*)d_in[9];
    const float* W2   = (const float*)d_in[10];
    const float* as2  = (const float*)d_in[11];
    const float* ad2  = (const float*)d_in[12];
    const float* b2   = (const float*)d_in[13];
    const float* ln2g = (const float*)d_in[14];
    const float* ln2b = (const float*)d_in[15];
    const float* W3   = (const float*)d_in[16];
    const float* as3  = (const float*)d_in[17];
    const float* ad3  = (const float*)d_in[18];
    const float* b3   = (const float*)d_in[19];
    const float* m1W  = (const float*)d_in[20];
    const float* m1b  = (const float*)d_in[21];
    const float* m2W  = (const float*)d_in[22];
    const float* m2b  = (const float*)d_in[23];
    const float* m3W  = (const float*)d_in[24];
    const float* m3b  = (const float*)d_in[25];

    // ---- carve workspace (all offsets 256B aligned) ----
    char* ws = (char*)d_ws;
    size_t off = 0;
    auto carve = [&](size_t bytes) -> char* {
        char* p = ws + off; off += (bytes + 255) & ~(size_t)255; return p;
    };
    float*    xl     = (float*)carve((size_t)NNODES * HC * 4);   // [N,256] (also MLP t2)
    float*    acc    = (float*)carve((size_t)NNODES * HC * 4);   // [N,256] message accum
    float*    h      = (float*)carve((size_t)NNODES * CH * 4);   // [N,64]
    float*    res    = (float*)carve((size_t)NNODES * CH * 4);   // [N,64] (also MLP t1)
    float*    es     = (float*)carve((size_t)NNODES * HEADS * 4);
    float*    ed     = (float*)carve((size_t)NNODES * HEADS * 4);
    unsigned* mseg   = (unsigned*)carve((size_t)NNODES * HEADS * 4);
    float*    sseg   = (float*)carve((size_t)NNODES * HEADS * 4);
    float*    ealpha = (float*)carve((size_t)EE * HEADS * 4);

    const int MT = NNODES / 16;   // 1875, exact

    // =================== layer 1: GATConv(128->64, H=4) ===================
    wmma_gemm<256, 128, 0, false><<<dim3(MT, 4), 128, 0, stream>>>(x, W1, nullptr, xl);
    wmma_gemm<64, 128, 0, true><<<dim3(MT, 1), 128, 0, stream>>>(x, resW, resb, res);
    attn_node<4><<<(NNODES * 4 + 255) / 256, 256, 0, stream>>>(xl, as1, ad1, es, ed);
    fill_u32<<<(NNODES * 4 + 255) / 256, 256, 0, stream>>>(mseg, ORD_NEG_INF, NNODES * 4);
    fill_f32<<<(NNODES * 4 + 255) / 256, 256, 0, stream>>>(sseg, 0.f, NNODES * 4);
    fill_f32<<<(NNODES * HC + 255) / 256, 256, 0, stream>>>(acc, 0.f, NNODES * HC);
    edge_max<4><<<(EE + 255) / 256, 256, 0, stream>>>(ei, es, ed, mseg);
    edge_sum<4><<<(EE + 255) / 256, 256, 0, stream>>>(ei, es, ed, mseg, sseg, ealpha);
    edge_msg<4><<<EE, 256, 0, stream>>>(ei, xl, ealpha, sseg, acc);
    mean_ln_res<<<(NNODES + 7) / 8, 256, 0, stream>>>(acc, b1, ln1g, ln1b, res, h);

    // =================== layer 2: GATConv(64->64, H=4), identity residual ==
    wmma_gemm<256, 64, 0, false><<<dim3(MT, 4), 128, 0, stream>>>(h, W2, nullptr, xl);
    attn_node<4><<<(NNODES * 4 + 255) / 256, 256, 0, stream>>>(xl, as2, ad2, es, ed);
    fill_u32<<<(NNODES * 4 + 255) / 256, 256, 0, stream>>>(mseg, ORD_NEG_INF, NNODES * 4);
    fill_f32<<<(NNODES * 4 + 255) / 256, 256, 0, stream>>>(sseg, 0.f, NNODES * 4);
    fill_f32<<<(NNODES * HC + 255) / 256, 256, 0, stream>>>(acc, 0.f, NNODES * HC);
    edge_max<4><<<(EE + 255) / 256, 256, 0, stream>>>(ei, es, ed, mseg);
    edge_sum<4><<<(EE + 255) / 256, 256, 0, stream>>>(ei, es, ed, mseg, sseg, ealpha);
    edge_msg<4><<<EE, 256, 0, stream>>>(ei, xl, ealpha, sseg, acc);
    mean_ln_res<<<(NNODES + 7) / 8, 256, 0, stream>>>(acc, b2, ln2g, ln2b, h, h); // in-place safe

    // =================== layer 3: GATConv(64->64, H=1) =====================
    wmma_gemm<64, 64, 0, false><<<dim3(MT, 1), 128, 0, stream>>>(h, W3, nullptr, xl);
    attn_node<1><<<(NNODES + 255) / 256, 256, 0, stream>>>(xl, as3, ad3, es, ed);
    fill_u32<<<(NNODES + 255) / 256, 256, 0, stream>>>(mseg, ORD_NEG_INF, NNODES);
    fill_f32<<<(NNODES + 255) / 256, 256, 0, stream>>>(sseg, 0.f, NNODES);
    fill_f32<<<(NNODES * CH + 255) / 256, 256, 0, stream>>>(acc, 0.f, NNODES * CH);
    edge_max<1><<<(EE + 255) / 256, 256, 0, stream>>>(ei, es, ed, mseg);
    edge_sum<1><<<(EE + 255) / 256, 256, 0, stream>>>(ei, es, ed, mseg, sseg, ealpha);
    edge_msg<1><<<EE, 64, 0, stream>>>(ei, xl, ealpha, sseg, acc);
    add_bias64<<<(NNODES * CH + 255) / 256, 256, 0, stream>>>(acc, b3, h);

    // =================== MLP 64->64->32->1 + sigmoid =======================
    wmma_gemm<64, 64, 1, true><<<dim3(MT, 1), 128, 0, stream>>>(h, m1W, m1b, res);   // ELU
    wmma_gemm<32, 64, 1, true><<<dim3(MT, 1), 128, 0, stream>>>(res, m2W, m2b, xl);  // ELU
    mlp_final<<<(NNODES + 255) / 256, 256, 0, stream>>>(xl, m3W, m3b, (float*)d_out);
}
